// LinearSelfAttention_29317446762883
// MI455X (gfx1250) — matrix operand order repaired and checked by reading, not compiled, not run on hardware
//
#include <hip/hip_runtime.h>
#include <hip/hip_bf16.h>
#include <stdint.h>

// ---------------- problem constants ----------------
#define BATCH  4
#define SEQ    8192
#define HEADS  16
#define HDIM   64
#define DMODEL 1024
#define FDIM   128
#define BH     (BATCH * HEADS)   // 64
#define MROWS  (BATCH * SEQ)     // 32768

// ---------------- bf16 / wmma helpers ----------------
typedef __bf16 bf16_t;
typedef __bf16 bf16x8  __attribute__((ext_vector_type(8)));
typedef __bf16 bf16x16 __attribute__((ext_vector_type(16)));
typedef float  f32x8   __attribute__((ext_vector_type(8)));

static __device__ __forceinline__ bf16_t f2bf(float f) {
  unsigned int u = __builtin_bit_cast(unsigned int, f);
  unsigned int r = u + 0x7FFFu + ((u >> 16) & 1u);   // round-to-nearest-even
  unsigned short h = (unsigned short)(r >> 16);
  return __builtin_bit_cast(bf16_t, h);
}
static __device__ __forceinline__ float bf2f(bf16_t b) {
  unsigned short h = __builtin_bit_cast(unsigned short, b);
  unsigned int u = ((unsigned int)h) << 16;
  return __builtin_bit_cast(float, u);
}
static __device__ __forceinline__ bf16x16 cat16(bf16x8 lo, bf16x8 hi) {
  return __builtin_shufflevector(lo, hi, 0,1,2,3,4,5,6,7,8,9,10,11,12,13,14,15);
}
static __device__ __forceinline__ f32x8 wmma_bf16(bf16x16 a, bf16x16 b, f32x8 c) {
  return __builtin_amdgcn_wmma_f32_16x16x32_bf16(false, a, false, b, (short)0, c, false, false);
}

// ---- CDNA5 async global->LDS copy (ASYNCcnt path, ISA 10.x / 15.18.3) ----
// Flat addresses in the LDS aperture map to LDS via addr[31:0] truncation
// (ISA 10.2), so the low 32 bits of a generic __shared__ pointer are the LDS
// byte address required in VDST.
static __device__ __forceinline__ void async_ld_b128(void* lds_p, const void* gp) {
  unsigned int       l = (unsigned int)(uintptr_t)lds_p;
  unsigned long long g = (unsigned long long)(uintptr_t)gp;
  asm volatile("global_load_async_to_lds_b128 %0, %1, off"
               :: "v"(l), "v"(g) : "memory");
}
static __device__ __forceinline__ void wait_async0() {
  asm volatile("s_wait_asynccnt 0x0" ::: "memory");
}

// ---------------- utility kernels ----------------
__global__ void cast_f32_to_bf16(const float* __restrict__ src, bf16_t* __restrict__ dst, int n) {
  int i = blockIdx.x * 256 + threadIdx.x;
  if (i < n) dst[i] = f2bf(src[i]);
}

__global__ void zero_f32(float* __restrict__ p, int n) {
  int i = blockIdx.x * 256 + threadIdx.x;
  if (i < n) p[i] = 0.0f;
}

// One-time weight transpose+downcast: W f32 [K,N] -> WT bf16 [N,K]
__global__ void __launch_bounds__(256)
transpose_cast_kernel(const float* __restrict__ W, bf16_t* __restrict__ WT) {
  __shared__ float tile[32][33];
  const int k0 = blockIdx.y * 32, n0 = blockIdx.x * 32;
  const int tx = threadIdx.x & 31, ty = threadIdx.x >> 5;  // 32 x 8
#pragma unroll
  for (int i = 0; i < 32; i += 8)
    tile[ty + i][tx] = W[(size_t)(k0 + ty + i) * DMODEL + n0 + tx];
  __syncthreads();
#pragma unroll
  for (int i = 0; i < 32; i += 8)
    WT[(size_t)(n0 + ty + i) * DMODEL + k0 + tx] = f2bf(tile[tx][ty + i]);
}

// ---------------- generic bf16 WMMA GEMM ----------------
// C[M,N] = A[M,K] (row-major bf16) * Bt[N,K] (PRE-TRANSPOSED bf16)
// 128x128 block tile, 8 waves (4 M x 2 N), wave tile 32x64 = 2x4 accums.
// Double-buffered LDS fed by async global->LDS b128 copies.
#define BM 128
#define BN 128
#define BK 32

__global__ void __launch_bounds__(256)
gemm_bf16_wmma(const bf16_t* __restrict__ A, const bf16_t* __restrict__ Bt,
               float* __restrict__ Cf, bf16_t* __restrict__ Cb,
               int M, int N, int K, int outBf16) {
  __shared__ __align__(16) bf16_t sA [2][BM][BK];  // 2 x 8 KB
  __shared__ __align__(16) bf16_t sBt[2][BN][BK];  // 2 x 8 KB  [n][k]

  const int tid  = threadIdx.x;
  const int lane = tid & 31;
  const int wave = tid >> 5;
  const int wM   = wave & 3;
  const int wN   = wave >> 2;
  const int blockM = blockIdx.y * BM;
  const int blockN = blockIdx.x * BN;

  f32x8 acc[2][4] = {};

  // both tiles are 128 rows x 32 k: 512 chunks of 8 bf16; 2 chunks / thread each
  auto load_tiles = [&](int k0, int buf) {
#pragma unroll
    for (int i = 0; i < 2; ++i) {
      int c  = tid + i * 256;
      int r  = c >> 2;
      int cc = (c & 3) * 8;
      async_ld_b128(&sA [buf][r][cc], A  + (size_t)(blockM + r) * K + k0 + cc);
      async_ld_b128(&sBt[buf][r][cc], Bt + (size_t)(blockN + r) * K + k0 + cc);
    }
  };

  load_tiles(0, 0);
  wait_async0();
  __syncthreads();

  const int rsel = lane & 15;
  const int kb   = (lane >> 4) * 8;    // A frag: lanes16-31 start K=8 (then +16)
  const int kk   = (lane >> 4) * 16;   // unused: B frag reads K at kb, kb+16 like A

  const int nIter = K / BK;
  for (int it = 0; it < nIter; ++it) {
    const int buf = it & 1;
    if (it + 1 < nIter) load_tiles((it + 1) * BK, buf ^ 1);  // overlap with WMMA

    bf16x16 af[2];
#pragma unroll
    for (int mi = 0; mi < 2; ++mi) {
      int m = wM * 32 + mi * 16 + rsel;
      af[mi] = cat16(*(const bf16x8*)(&sA[buf][m][kb]),
                     *(const bf16x8*)(&sA[buf][m][kb + 16]));
    }
    bf16x16 bfv[4];
#pragma unroll
    for (int ni = 0; ni < 4; ++ni) {
      int n = wN * 64 + ni * 16 + rsel;
      bfv[ni] = cat16(*(const bf16x8*)(&sBt[buf][n][kb]),
                      *(const bf16x8*)(&sBt[buf][n][kb + 16]));
    }
#pragma unroll
    for (int mi = 0; mi < 2; ++mi)
#pragma unroll
      for (int ni = 0; ni < 4; ++ni)
        acc[mi][ni] = wmma_bf16(af[mi], bfv[ni], acc[mi][ni]);

    wait_async0();     // own prefetch landed in other buffer
    __syncthreads();   // everyone done reading this buffer
  }

  const int n0 = lane & 15;
  const int mb = (lane >> 4) * 8;
#pragma unroll
  for (int mi = 0; mi < 2; ++mi)
#pragma unroll
    for (int ni = 0; ni < 4; ++ni)
#pragma unroll
      for (int v = 0; v < 8; ++v) {
        size_t row = (size_t)(blockM + wM * 32 + mi * 16 + mb + v);
        size_t col = (size_t)(blockN + wN * 64 + ni * 16 + n0);
        float val = acc[mi][ni][v];
        if (outBf16) Cb[row * (size_t)N + col] = f2bf(val);
        else         Cf[row * (size_t)N + col] = val;
      }
}

// ---------------- phi feature map ----------------
__global__ void __launch_bounds__(128)
phi_kernel(const bf16_t* __restrict__ X, const float* __restrict__ rf,
           bf16_t* __restrict__ P, float* __restrict__ ksum, int doSum) {
  __shared__ float s_rf[HDIM * FDIM];   // 32 KB
  __shared__ float qrow[HDIM];

  const int bh = blockIdx.x;
  const int b  = bh >> 4;
  const int h  = bh & 15;
  const int s0 = blockIdx.y * 64;
  const int f  = threadIdx.x;

  for (int i = threadIdx.x; i < HDIM * FDIM; i += 128) s_rf[i] = rf[i];
  __syncthreads();

  const float invs = 0.08838834764831845f;  // 128^-0.5
  float fsum = 0.0f;
  for (int si = 0; si < 64; ++si) {
    int s = s0 + si;
    if (threadIdx.x < HDIM)
      qrow[threadIdx.x] = bf2f(X[((size_t)b * SEQ + s) * DMODEL + h * HDIM + threadIdx.x]);
    __syncthreads();
    float acc = 0.0f;
#pragma unroll 8
    for (int d = 0; d < HDIM; ++d) acc += qrow[d] * s_rf[d * FDIM + f];
    float val = ((f < 64) ? __cosf(acc) : __sinf(acc)) * invs;
    P[((size_t)bh * SEQ + s) * FDIM + f] = f2bf(val);
    fsum += val;
    __syncthreads();
  }
  if (doSum) atomicAdd(&ksum[bh * FDIM + f], fsum);
}

// ---------------- KV = Kp^T * V per (b,h), WMMA, fp32 atomic accumulate ----
#define KV_SLAB 512
__global__ void __launch_bounds__(256)
kv_kernel(const bf16_t* __restrict__ Kp, const bf16_t* __restrict__ V,
          float* __restrict__ KV) {
  __shared__ __align__(16) bf16_t sKt[FDIM][32];  // Kp^T tile [f][s]
  __shared__ __align__(16) bf16_t sVt[HDIM][32];  // V^T  tile [d][s]

  const int bh = blockIdx.x;
  const int b  = bh >> 4;
  const int h  = bh & 15;
  const int tid = threadIdx.x, lane = tid & 31, wave = tid >> 5;
  const int wM = wave & 3;
  const int wN = wave >> 2;

  const bf16_t* KpBase = Kp + (size_t)bh * SEQ * FDIM;
  const bf16_t* VBase  = V  + (size_t)b * SEQ * DMODEL + h * HDIM;

  f32x8 acc[2][2] = {};
  const int sBeg = blockIdx.y * KV_SLAB;

  for (int s0 = sBeg; s0 < sBeg + KV_SLAB; s0 += 32) {
#pragma unroll
    for (int i = 0; i < 2; ++i) {
      int c  = tid + i * 256;
      int r  = c >> 4;
      int fc = (c & 15) * 8;
      bf16x8 v = *(const bf16x8*)(KpBase + (size_t)(s0 + r) * FDIM + fc);
#pragma unroll
      for (int e = 0; e < 8; ++e) sKt[fc + e][r] = v[e];
    }
    {
      int r  = tid >> 3;
      int dc = (tid & 7) * 8;
      bf16x8 v = *(const bf16x8*)(VBase + (size_t)(s0 + r) * DMODEL + dc);
#pragma unroll
      for (int e = 0; e < 8; ++e) sVt[dc + e][r] = v[e];
    }
    __syncthreads();

    const int rsel = lane & 15;
    const int kb = (lane >> 4) * 8;
    const int kk = (lane >> 4) * 16;
    bf16x16 af[2], bfv[2];
#pragma unroll
    for (int mi = 0; mi < 2; ++mi) {
      int m = wM * 32 + mi * 16 + rsel;
      af[mi] = cat16(*(const bf16x8*)(&sKt[m][kb]), *(const bf16x8*)(&sKt[m][kb + 16]));
    }
#pragma unroll
    for (int ni = 0; ni < 2; ++ni) {
      int n = wN * 32 + ni * 16 + rsel;
      bfv[ni] = cat16(*(const bf16x8*)(&sVt[n][kb]), *(const bf16x8*)(&sVt[n][kb + 16]));
    }
#pragma unroll
    for (int mi = 0; mi < 2; ++mi)
#pragma unroll
      for (int ni = 0; ni < 2; ++ni)
        acc[mi][ni] = wmma_bf16(af[mi], bfv[ni], acc[mi][ni]);
    __syncthreads();
  }

  const int n0 = lane & 15;
  const int mb = (lane >> 4) * 8;
#pragma unroll
  for (int mi = 0; mi < 2; ++mi)
#pragma unroll
    for (int ni = 0; ni < 2; ++ni)
#pragma unroll
      for (int v = 0; v < 8; ++v) {
        int fr = wM * 32 + mi * 16 + mb + v;
        int dc = wN * 32 + ni * 16 + n0;
        atomicAdd(&KV[((size_t)bh * FDIM + fr) * HDIM + dc], acc[mi][ni][v]);
      }
}

// KV f32 [bh,128f,64d] -> KVbT bf16 [bh,64d,128f]
__global__ void kvcast_kernel(const float* __restrict__ KV, bf16_t* __restrict__ KVbT) {
  int bh = blockIdx.x;
  for (int i = threadIdx.x; i < FDIM * HDIM; i += 256) {
    int fr = i >> 6, dc = i & 63;
    KVbT[((size_t)bh * HDIM + dc) * FDIM + fr] = f2bf(KV[(size_t)bh * FDIM * HDIM + i]);
  }
}

// Zinv[bh*S+s] = 1 / max(Qp[s,:]·ksum, 1e-6)
__global__ void __launch_bounds__(256)
zinv_kernel(const bf16_t* __restrict__ Qp, const float* __restrict__ ksum,
            float* __restrict__ zinv) {
  size_t idx = (size_t)blockIdx.x * 256 + threadIdx.x;
  int bh = (int)(idx >> 13);
  const bf16_t* q = Qp + idx * FDIM;
  const float* ks = ksum + bh * FDIM;
  float z = 0.0f;
#pragma unroll 8
  for (int f = 0; f < FDIM; ++f) z += bf2f(q[f]) * ks[f];
  z = fmaxf(z, 1e-6f);
  zinv[idx] = 1.0f / z;
}

// ---------------- attn = (Qp @ KV) * Zinv, WMMA ----------------
__global__ void __launch_bounds__(256)
qkv_kernel(const bf16_t* __restrict__ Qp, const bf16_t* __restrict__ KVbT,
           const float* __restrict__ zinv, bf16_t* __restrict__ attn) {
  __shared__ __align__(16) bf16_t sKV[HDIM * FDIM];  // [d][f] 16 KB

  const int bh = blockIdx.x;
  const int b  = bh >> 4;
  const int h  = bh & 15;
  const int tid = threadIdx.x, lane = tid & 31, wave = tid >> 5;

  for (int i = tid * 8; i < HDIM * FDIM; i += 256 * 8)
    async_ld_b128(sKV + i, KVbT + (size_t)bh * HDIM * FDIM + i);
  wait_async0();
  __syncthreads();

  const int s0 = blockIdx.y * 256 + wave * 32;
  const bf16_t* Qb = Qp + ((size_t)bh * SEQ + s0) * FDIM;

  f32x8 acc[2][4] = {};
  const int rsel = lane & 15;
  const int kb = (lane >> 4) * 8;
  const int kk = (lane >> 4) * 16;

#pragma unroll
  for (int k0 = 0; k0 < FDIM; k0 += 32) {
    bf16x16 af[2];
#pragma unroll
    for (int mi = 0; mi < 2; ++mi) {
      const bf16_t* p = Qb + (size_t)(mi * 16 + rsel) * FDIM + k0;
      af[mi] = cat16(*(const bf16x8*)(p + kb), *(const bf16x8*)(p + kb + 16));
    }
#pragma unroll
    for (int ni = 0; ni < 4; ++ni) {
      const bf16_t* p = sKV + (ni * 16 + rsel) * FDIM + k0;
      bf16x16 bf = cat16(*(const bf16x8*)(p + kb), *(const bf16x8*)(p + kb + 16));
#pragma unroll
      for (int mi = 0; mi < 2; ++mi)
        acc[mi][ni] = wmma_bf16(af[mi], bf, acc[mi][ni]);
    }
  }

  const int n0 = lane & 15;
  const int mb = (lane >> 4) * 8;
#pragma unroll
  for (int mi = 0; mi < 2; ++mi)
#pragma unroll
    for (int v = 0; v < 8; ++v) {
      int s = s0 + mi * 16 + mb + v;
      float zi = zinv[(size_t)bh * SEQ + s];
#pragma unroll
      for (int ni = 0; ni < 4; ++ni) {
        size_t col = (size_t)h * HDIM + ni * 16 + n0;
        attn[((size_t)b * SEQ + s) * DMODEL + col] = f2bf(acc[mi][ni][v] * zi);
      }
    }
}

// ---------------- host launcher ----------------
extern "C" void kernel_launch(void* const* d_in, const int* in_sizes, int n_in,
                              void* d_out, int out_size, void* d_ws, size_t ws_size,
                              hipStream_t stream) {
  (void)in_sizes; (void)n_in; (void)out_size; (void)ws_size;
  const float* x  = (const float*)d_in[0];
  const float* Wq = (const float*)d_in[1];
  const float* Wk = (const float*)d_in[2];
  const float* Wv = (const float*)d_in[3];
  const float* Wo = (const float*)d_in[4];
  const float* rf = (const float*)d_in[5];
  float* out = (float*)d_out;

  char* ws = (char*)d_ws;
  size_t off = 0;
  auto alloc = [&](size_t bytes) -> char* {
    char* p = ws + off;
    off += (bytes + 255) & ~(size_t)255;
    return p;
  };
  bf16_t* xb   = (bf16_t*)alloc((size_t)MROWS * DMODEL * 2);   // 64 MB
  bf16_t* WqT  = (bf16_t*)alloc((size_t)DMODEL * DMODEL * 2);  // 2 MB (transposed)
  bf16_t* WkT  = (bf16_t*)alloc((size_t)DMODEL * DMODEL * 2);
  bf16_t* WvT  = (bf16_t*)alloc((size_t)DMODEL * DMODEL * 2);
  bf16_t* WoT  = (bf16_t*)alloc((size_t)DMODEL * DMODEL * 2);
  bf16_t* Qb   = (bf16_t*)alloc((size_t)MROWS * DMODEL * 2);   // 64 MB (reused as attn)
  bf16_t* Kb   = (bf16_t*)alloc((size_t)MROWS * DMODEL * 2);
  bf16_t* Vb   = (bf16_t*)alloc((size_t)MROWS * DMODEL * 2);
  bf16_t* Qp   = (bf16_t*)alloc((size_t)BH * SEQ * FDIM * 2);  // 128 MB
  bf16_t* Kp   = (bf16_t*)alloc((size_t)BH * SEQ * FDIM * 2);
  float*  Ksum = (float*)alloc((size_t)BH * FDIM * 4);
  float*  KV   = (float*)alloc((size_t)BH * FDIM * HDIM * 4);
  bf16_t* KVbT = (bf16_t*)alloc((size_t)BH * HDIM * FDIM * 2);
  float*  Zinv = (float*)alloc((size_t)BH * SEQ * 4);
  bf16_t* attn = Qb;

  // 1) downcast x; transpose+downcast weights (one-time, makes GEMM B-tiles
  //    contiguous so all staging is async b128 direct-to-LDS)
  {
    int n = MROWS * DMODEL;
    cast_f32_to_bf16<<<(n + 255) / 256, 256, 0, stream>>>(x, xb, n);
    dim3 gT(DMODEL / 32, DMODEL / 32);
    transpose_cast_kernel<<<gT, 256, 0, stream>>>(Wq, WqT);
    transpose_cast_kernel<<<gT, 256, 0, stream>>>(Wk, WkT);
    transpose_cast_kernel<<<gT, 256, 0, stream>>>(Wv, WvT);
    transpose_cast_kernel<<<gT, 256, 0, stream>>>(Wo, WoT);
  }

  // 2) projection GEMMs (WMMA, async double-buffered)
  dim3 gGemm(DMODEL / BN, MROWS / BM);  // (8, 256)
  gemm_bf16_wmma<<<gGemm, 256, 0, stream>>>(xb, WqT, nullptr, Qb, MROWS, DMODEL, DMODEL, 1);
  gemm_bf16_wmma<<<gGemm, 256, 0, stream>>>(xb, WkT, nullptr, Kb, MROWS, DMODEL, DMODEL, 1);
  gemm_bf16_wmma<<<gGemm, 256, 0, stream>>>(xb, WvT, nullptr, Vb, MROWS, DMODEL, DMODEL, 1);

  // 3) zero accumulators
  {
    int n1 = BH * FDIM * HDIM;
    zero_f32<<<(n1 + 255) / 256, 256, 0, stream>>>(KV, n1);
    int n2 = BH * FDIM;
    zero_f32<<<(n2 + 255) / 256, 256, 0, stream>>>(Ksum, n2);
  }

  // 4) phi feature maps (+ Kp column sum)
  dim3 gPhi(BH, SEQ / 64);
  phi_kernel<<<gPhi, 128, 0, stream>>>(Qb, rf, Qp, Ksum, 0);
  phi_kernel<<<gPhi, 128, 0, stream>>>(Kb, rf, Kp, Ksum, 1);

  // 5) KV = Kp^T @ V (WMMA + f32 atomics)
  dim3 gKV(BH, SEQ / KV_SLAB);  // (64, 16)
  kv_kernel<<<gKV, 256, 0, stream>>>(Kp, Vb, KV);
  kvcast_kernel<<<BH, 256, 0, stream>>>(KV, KVbT);

  // 6) Z normalizer
  zinv_kernel<<<(BH * SEQ) / 256, 256, 0, stream>>>(Qp, Ksum, Zinv);

  // 7) attn = (Qp @ KV) / Z (WMMA)
  dim3 gQKV(BH, SEQ / 256);  // (64, 32)
  qkv_kernel<<<gQKV, 256, 0, stream>>>(Qp, KVbT, Zinv, attn);

  // 8) out = attn @ Wo (WMMA, f32 out)
  gemm_bf16_wmma<<<gGemm, 256, 0, stream>>>(attn, WoT, out, nullptr, MROWS, DMODEL, DMODEL, 0);
}
